// MoeMuxExpertChoiceKTokens_40123584479377
// MI455X (gfx1250) — matrix operand order repaired and checked by reading, not compiled, not run on hardware
//
#include <hip/hip_runtime.h>

#define BB 512
#define SS 2048
#define DD 128
#define EE 64
#define HH 512
#define OO 128
#define KK 8

typedef __attribute__((ext_vector_type(16))) __bf16 v16bf;
typedef __attribute__((ext_vector_type(8)))  __bf16 v8bf;
typedef __attribute__((ext_vector_type(8)))  float  v8f;

// ---------------------------------------------------------------- zero output
__global__ __launch_bounds__(256) void k_zero(float* __restrict__ out, size_t n4) {
    size_t i = (size_t)blockIdx.x * blockDim.x + threadIdx.x;
    size_t stride = (size_t)gridDim.x * blockDim.x;
    float4 z = make_float4(0.f, 0.f, 0.f, 0.f);
    for (; i < n4; i += stride) ((float4*)out)[i] = z;
}

// ------------------------------------------------- convert expert weights bf16
__global__ __launch_bounds__(256) void k_cvt(const float* __restrict__ w1,
                                             const float* __restrict__ w2,
                                             unsigned short* __restrict__ w1b_,
                                             unsigned short* __restrict__ w2b_) {
    __bf16* w1b = (__bf16*)w1b_;
    __bf16* w2b = (__bf16*)w2b_;
    const size_t n1 = (size_t)EE * (DD + 1) * HH;
    const size_t n2 = (size_t)EE * (HH + 1) * OO;
    size_t i = (size_t)blockIdx.x * blockDim.x + threadIdx.x;
    size_t stride = (size_t)gridDim.x * blockDim.x;
    for (; i < n1 + n2; i += stride) {
        if (i < n1) w1b[i] = (__bf16)w1[i];
        else        w2b[i - n1] = (__bf16)w2[i - n1];
    }
}

// --------- fused gate GEMM (WMMA bf16) + online softmax stats + top-8 per expert
// one block per batch b; 8 waves each produce a 16-token x 64-expert logit tile.
// gate weights are staged in LDS in WMMA-fragment-native layout:
//   s_gw[((kk*4+nt)*32 + lane)*16 + j]  ==  B[k][e],  k = kk*32 + (lane>>4)*16 + j,
//                                                     e = nt*16 + (lane&15)
__global__ __launch_bounds__(256) void k_gate(const float* __restrict__ x,
                                              const float* __restrict__ gw,
                                              const float* __restrict__ gb,
                                              float* __restrict__ topv,
                                              int* __restrict__ topi) {
    __shared__ alignas(32) __bf16 s_gw[4 * 4 * 32 * 16]; // 16 KB, fragment layout
    __shared__ float s_gb[EE];
    __shared__ float s_log[128 * EE];                    // 32 KB

    const int b   = blockIdx.x;
    const int tid = threadIdx.x;
    const int w    = tid >> 5;
    const int lane = tid & 31;
    const int half = lane >> 4;
    const int l16  = lane & 15;

    for (int i = tid; i < 4 * 4 * 32 * 16; i += 256) {
        const int j  = i & 15;
        const int ln = (i >> 4) & 31;
        const int nt = (i >> 9) & 3;
        const int kk = i >> 11;
        const int k  = kk * 32 + (ln >> 4) * 16 + j;
        const int e  = nt * 16 + (ln & 15);
        s_gw[i] = (__bf16)gw[e * DD + k];
    }
    if (tid < EE) s_gb[tid] = gb[tid];
    __syncthreads();

    const v16bf* s_gwv = (const v16bf*)s_gw;

    // per-expert top-8 state (threads 0..63)
    float m = -1e30f, l = 0.f;
    float tv[KK]; int ti[KK];
#pragma unroll
    for (int k = 0; k < KK; ++k) { tv[k] = -1e30f; ti[k] = 0; }

    for (int chunk = 0; chunk < SS / 128; ++chunk) {
        const int s0 = chunk * 128;
        const float* xrow = x + ((size_t)b * SS + s0 + w * 16 + l16) * DD;

        v8f c[4] = {};
#pragma unroll
        for (int kk = 0; kk < DD / 32; ++kk) {
            const int k0 = kk * 32;
            // A fragment: one contiguous 16-float run (k = half*16 + j) -> 4x b128 + pk cvt
            const float4 f0 = *(const float4*)(xrow + k0 + half * 16);
            const float4 f1 = *(const float4*)(xrow + k0 + half * 16 + 4);
            const float4 f2 = *(const float4*)(xrow + k0 + half * 16 + 8);
            const float4 f3 = *(const float4*)(xrow + k0 + half * 16 + 12);
            v16bf a;
            a[0]  = (__bf16)f0.x; a[1]  = (__bf16)f0.y; a[2]  = (__bf16)f0.z; a[3]  = (__bf16)f0.w;
            a[4]  = (__bf16)f1.x; a[5]  = (__bf16)f1.y; a[6]  = (__bf16)f1.z; a[7]  = (__bf16)f1.w;
            a[8]  = (__bf16)f2.x; a[9]  = (__bf16)f2.y; a[10] = (__bf16)f2.z; a[11] = (__bf16)f2.w;
            a[12] = (__bf16)f3.x; a[13] = (__bf16)f3.y; a[14] = (__bf16)f3.z; a[15] = (__bf16)f3.w;
#pragma unroll
            for (int nt = 0; nt < 4; ++nt) {
                const v16bf bb = s_gwv[(kk * 4 + nt) * 32 + lane];
                c[nt] = __builtin_amdgcn_wmma_f32_16x16x32_bf16(
                    false, a, false, bb, (short)0, c[nt], false, false);
            }
        }
        // bias + stage logits in LDS
#pragma unroll
        for (int nt = 0; nt < 4; ++nt) {
            const int e = nt * 16 + l16;
            const float bias = s_gb[e];
#pragma unroll
            for (int r = 0; r < 8; ++r) {
                const int mrow = w * 16 + r + half * 8;
                s_log[mrow * EE + e] = c[nt][r] + bias;
            }
        }
        __syncthreads();

        if (tid < EE) {
#pragma unroll 4
            for (int i = 0; i < 128; ++i) {
                const float v = s_log[i * EE + tid];
                // online softmax (branchless)
                const float mn = fmaxf(m, v);
                l = l * __expf(m - mn) + __expf(v - mn);
                m = mn;
                // branchless unrolled top-8 insertion (sorted descending)
                if (v > tv[KK - 1]) {
                    const int s = s0 + i;
                    float pv = 0.f; int pi_ = 0; bool pc = false;
#pragma unroll
                    for (int j = 0; j < KK; ++j) {
                        const bool  c2 = v > tv[j];
                        const float cv = tv[j]; const int ci = ti[j];
                        if (c2) { tv[j] = pc ? pv : v; ti[j] = pc ? pi_ : s; }
                        pv = cv; pi_ = ci; pc = c2;
                    }
                }
            }
        }
        __syncthreads();
    }

    if (tid < EE) {
        const float inv = 1.f / l;
#pragma unroll
        for (int k = 0; k < KK; ++k) {
            topv[((size_t)b * EE + tid) * KK + k] = __expf(tv[k] - m) * inv;
            topi[((size_t)b * EE + tid) * KK + k] = ti[k];
        }
    }
}

// ------------------------------------------- gather + weighted pool -> bf16 [E][B][D]
__global__ __launch_bounds__(128) void k_pool(const float* __restrict__ x,
                                              const float* __restrict__ topv,
                                              const int* __restrict__ topi,
                                              unsigned short* __restrict__ pooled_) {
    __bf16* pooled = (__bf16*)pooled_;
    const int b = blockIdx.x >> 6, e = blockIdx.x & 63, d = threadIdx.x;
    const float* tvp = topv + ((size_t)b * EE + e) * KK;
    const int*   tip = topi + ((size_t)b * EE + e) * KK;
    float acc = 0.f;
#pragma unroll
    for (int k = 0; k < KK; ++k)
        acc += tvp[k] * x[((size_t)b * SS + tip[k]) * DD + d];
    pooled[((size_t)e * BB + b) * DD + d] = (__bf16)acc;
}

// A fragment from row-major bf16 (one contiguous 32B run, k = half*16 + j) -> 2x b128
__device__ __forceinline__ v16bf load_a_frag(const __bf16* arow, int k0, int half) {
    const v8bf lo = *(const v8bf*)(arow + k0 + half * 16);
    const v8bf hi = *(const v8bf*)(arow + k0 + half * 16 + 8);
    return __builtin_shufflevector(lo, hi, 0, 1, 2, 3, 4, 5, 6, 7,
                                           8, 9, 10, 11, 12, 13, 14, 15);
}

// stage a 128(K) x 64(N) bf16 tile from row-major global into fragment layout
__device__ __forceinline__ void stage_b_tile(__bf16* s_b, const __bf16* src,
                                             int ldb, int n0, int tid) {
    for (int i = tid; i < 128 * 64; i += 256) {
        const int k = i >> 6, nc = i & 63;
        const int g = ((k >> 5) * 4 + (nc >> 4)) * 32 + ((k >> 4) & 1) * 16 + (nc & 15);
        s_b[g * 16 + (k & 15)] = src[(size_t)k * ldb + n0 + nc];
    }
}

// ------------------------------ expert GEMM1: h = gelu(pooled @ W1 + b1), WMMA bf16
__global__ __launch_bounds__(256) void k_mlp1(const unsigned short* __restrict__ pooled_,
                                              const unsigned short* __restrict__ w1b_,
                                              const float* __restrict__ w1f,
                                              unsigned short* __restrict__ hmat_) {
    const __bf16* pooled = (const __bf16*)pooled_;
    const __bf16* w1b    = (const __bf16*)w1b_;
    __bf16*       hmat   = (__bf16*)hmat_;
    __shared__ alignas(32) __bf16 s_b[128 * 64]; // 16 KB fragment-layout W tile

    const int blk = blockIdx.x;
    const int e  = blk >> 6;
    const int b0 = ((blk >> 3) & 7) * 64;
    const int h0 = (blk & 7) * 64;
    const int tid = threadIdx.x;
    const int w = tid >> 5, lane = tid & 31, half = lane >> 4, l16 = lane & 15;
    const int r0 = (w & 3) * 16;
    const int c0 = (w >> 2) * 32;

    const __bf16* A  = pooled + (size_t)e * BB * DD;
    const __bf16* Bm = w1b + (size_t)e * (DD + 1) * HH;

    stage_b_tile(s_b, Bm, HH, h0, tid);
    __syncthreads();
    const v16bf* sbv = (const v16bf*)s_b;

    const __bf16* arow = A + (size_t)(b0 + r0 + l16) * DD;
    v8f c[2] = {};
#pragma unroll
    for (int kk = 0; kk < DD / 32; ++kk) {
        const v16bf a = load_a_frag(arow, kk * 32, half);
#pragma unroll
        for (int nt = 0; nt < 2; ++nt) {
            const int nt4 = (c0 >> 4) + nt;
            const v16bf bb = sbv[(kk * 4 + nt4) * 32 + lane];
            c[nt] = __builtin_amdgcn_wmma_f32_16x16x32_bf16(
                false, a, false, bb, (short)0, c[nt], false, false);
        }
    }
#pragma unroll
    for (int nt = 0; nt < 2; ++nt) {
        const int n = h0 + c0 + nt * 16 + l16;
        const float bias = w1f[((size_t)e * (DD + 1) + DD) * HH + n];
#pragma unroll
        for (int r = 0; r < 8; ++r) {
            const int mrow = b0 + r0 + r + half * 8;
            const float v = c[nt][r] + bias;
            const float g = 0.5f * v * (1.0f + erff(v * 0.70710678118654752f)); // exact gelu
            hmat[((size_t)e * BB + mrow) * HH + n] = (__bf16)g;
        }
    }
}

// ---------------------------- expert GEMM2: out = h @ W2 + b2, WMMA bf16 -> f32 [B][E][O]
__global__ __launch_bounds__(256) void k_mlp2(const unsigned short* __restrict__ hmat_,
                                              const unsigned short* __restrict__ w2b_,
                                              const float* __restrict__ w2f,
                                              float* __restrict__ oute) {
    const __bf16* hmat = (const __bf16*)hmat_;
    const __bf16* w2b  = (const __bf16*)w2b_;
    __shared__ alignas(32) __bf16 s_b[128 * 64]; // 16 KB, one 128-K chunk at a time

    const int blk = blockIdx.x;
    const int e  = blk >> 4;
    const int b0 = ((blk >> 1) & 7) * 64;
    const int o0 = (blk & 1) * 64;
    const int tid = threadIdx.x;
    const int w = tid >> 5, lane = tid & 31, half = lane >> 4, l16 = lane & 15;
    const int r0 = (w & 3) * 16;
    const int c0 = (w >> 2) * 32;

    const __bf16* A  = hmat + (size_t)e * BB * HH;
    const __bf16* Bm = w2b + (size_t)e * (HH + 1) * OO;
    const __bf16* arow = A + (size_t)(b0 + r0 + l16) * HH;

    v8f c[2] = {};
    for (int kc = 0; kc < HH / 128; ++kc) {
        stage_b_tile(s_b, Bm + (size_t)kc * 128 * OO, OO, o0, tid);
        __syncthreads();
        const v16bf* sbv = (const v16bf*)s_b;
#pragma unroll
        for (int kk = 0; kk < 4; ++kk) {
            const v16bf a = load_a_frag(arow, kc * 128 + kk * 32, half);
#pragma unroll
            for (int nt = 0; nt < 2; ++nt) {
                const int nt4 = (c0 >> 4) + nt;
                const v16bf bb = sbv[(kk * 4 + nt4) * 32 + lane];
                c[nt] = __builtin_amdgcn_wmma_f32_16x16x32_bf16(
                    false, a, false, bb, (short)0, c[nt], false, false);
            }
        }
        __syncthreads();
    }
#pragma unroll
    for (int nt = 0; nt < 2; ++nt) {
        const int n = o0 + c0 + nt * 16 + l16;
        const float bias = w2f[((size_t)e * (HH + 1) + HH) * OO + n];
#pragma unroll
        for (int r = 0; r < 8; ++r) {
            const int mrow = b0 + r0 + r + half * 8;
            oute[((size_t)mrow * EE + e) * OO + n] = c[nt][r] + bias;
        }
    }
}

// --------------------------- deterministic scatter-add (one block owns batch b)
__global__ __launch_bounds__(128) void k_scatter(const float* __restrict__ oute,
                                                 const float* __restrict__ topv,
                                                 const int* __restrict__ topi,
                                                 float* __restrict__ out) {
    const int b = blockIdx.x, o = threadIdx.x;
    for (int e = 0; e < EE; ++e) {
        const float base = oute[((size_t)b * EE + e) * OO + o];
        const float* tvp = topv + ((size_t)b * EE + e) * KK;
        const int*   tip = topi + ((size_t)b * EE + e) * KK;
#pragma unroll
        for (int k = 0; k < KK; ++k) {
            float* p = out + ((size_t)b * SS + tip[k]) * OO + o;
            *p += tvp[k] * base;   // only this block/thread touches (b, *, o)
        }
    }
}

extern "C" void kernel_launch(void* const* d_in, const int* in_sizes, int n_in,
                              void* d_out, int out_size, void* d_ws, size_t ws_size,
                              hipStream_t stream) {
    (void)in_sizes; (void)n_in; (void)out_size; (void)ws_size;
    const float* x       = (const float*)d_in[0];
    const float* gate_w  = (const float*)d_in[1];
    const float* gate_b  = (const float*)d_in[2];
    const float* weight1 = (const float*)d_in[3];
    const float* weight2 = (const float*)d_in[4];
    float* out = (float*)d_out;

    size_t off = 0;
    auto take = [&](size_t bytes) -> void* {
        void* p = (void*)((char*)d_ws + off);
        off = (off + bytes + 255) & ~((size_t)255);
        return p;
    };
    float*          topv   = (float*)         take((size_t)BB * EE * KK * 4);
    int*            topi   = (int*)           take((size_t)BB * EE * KK * 4);
    unsigned short* pooled = (unsigned short*)take((size_t)EE * BB * DD * 2);
    unsigned short* w1b    = (unsigned short*)take((size_t)EE * (DD + 1) * HH * 2);
    unsigned short* w2b    = (unsigned short*)take((size_t)EE * (HH + 1) * OO * 2);
    unsigned short* hmat   = (unsigned short*)take((size_t)EE * BB * HH * 2);
    float*          oute   = (float*)         take((size_t)BB * EE * OO * 4);

    const size_t n4 = (size_t)BB * SS * OO / 4;
    k_zero<<<8192, 256, 0, stream>>>(out, n4);
    k_cvt<<<4096, 256, 0, stream>>>(weight1, weight2, w1b, w2b);
    k_gate<<<BB, 256, 0, stream>>>(x, gate_w, gate_b, topv, topi);
    k_pool<<<BB * EE, 128, 0, stream>>>(x, topv, topi, pooled);
    k_mlp1<<<EE * 64, 256, 0, stream>>>(pooled, w1b, weight1, hmat);
    k_mlp2<<<EE * 16, 256, 0, stream>>>(hmat, w2b, weight2, oute);
    k_scatter<<<BB, 128, 0, stream>>>(oute, topv, topi, out);
}
